// EdgeEmbedder_25305947308321
// MI455X (gfx1250) — matrix operand order repaired and checked by reading, not compiled, not run on hardware
//
#include <hip/hip_runtime.h>
#include <math.h>

#define C_S      256
#define C_P      128
#define FEAT_DIM 64
#define NUM_BINS 22
#define TOT_W1   236   // W1 row stride
#define PI_F     3.14159265358979323846f

typedef __attribute__((ext_vector_type(16))) _Float16 v16h;
typedef __attribute__((ext_vector_type(8)))  float    v8f;

union H8 { uint4 u4; _Float16 h[8]; };
union H4 { uint2 u2; _Float16 h[4]; };

// ---------------------------------------------------------------------------
// Precompute 1: p = s @ W_sp.T + b_sp ; A_row = p @ W1[:,0:64].T + b1 ;
//               A_col = p @ W1[:,64:128].T        (one block per node)
// ---------------------------------------------------------------------------
__global__ __launch_bounds__(256) void node_embed_kernel(
    const float* __restrict__ s,
    const float* __restrict__ W_sp, const float* __restrict__ b_sp,
    const float* __restrict__ W1,   const float* __restrict__ b1,
    float* __restrict__ Arow, float* __restrict__ Acol)
{
    __shared__ float srow[C_S];
    __shared__ float p[FEAT_DIM];
    const int node = blockIdx.x;
    const int tid  = threadIdx.x;

    srow[tid] = s[node * C_S + tid];
    __syncthreads();

    if (tid < FEAT_DIM) {
        float acc = b_sp[tid];
        const float* wr = W_sp + tid * C_S;
        for (int k = 0; k < C_S; ++k) acc += srow[k] * wr[k];
        p[tid] = acc;
    }
    __syncthreads();

    if (tid < C_P) {
        const int o = tid;
        float acc = b1[o];                       // fold b1 into A_row
        const float* wr = W1 + o * TOT_W1;       // cols 0:64
        #pragma unroll 8
        for (int f = 0; f < FEAT_DIM; ++f) acc += p[f] * wr[f];
        Arow[node * C_P + o] = acc;
    } else {
        const int o = tid - C_P;
        float acc = 0.0f;
        const float* wr = W1 + o * TOT_W1 + FEAT_DIM; // cols 64:128
        #pragma unroll 8
        for (int f = 0; f < FEAT_DIM; ++f) acc += p[f] * wr[f];
        Acol[node * C_P + o] = acc;
    }
}

// ---------------------------------------------------------------------------
// Precompute 2: Rtab[r] = (sincos_emb(r-(N-1)) @ W_rp.T + b_rp) @ W1[:,128:192].T
// one block per relative offset r in [0, 2N-2]
// ---------------------------------------------------------------------------
__global__ __launch_bounds__(128) void relpos_kernel(
    const float* __restrict__ W_rp, const float* __restrict__ b_rp,
    const float* __restrict__ W1,
    float* __restrict__ Rtab, int N)
{
    __shared__ float emb[FEAT_DIM];
    __shared__ float rp[FEAT_DIM];
    const int r = blockIdx.x;
    const float d = (float)(r - (N - 1));
    const int tid = threadIdx.x;

    if (tid < FEAT_DIM) {
        const int k = tid & 31;
        const float ang = d * PI_F / powf(2056.0f, (float)(2 * k) / 64.0f);
        emb[tid] = (tid < 32) ? sinf(ang) : cosf(ang);
    }
    __syncthreads();
    if (tid < FEAT_DIM) {
        float acc = b_rp[tid];
        const float* wr = W_rp + tid * FEAT_DIM;
        #pragma unroll 8
        for (int k = 0; k < FEAT_DIM; ++k) acc += emb[k] * wr[k];
        rp[tid] = acc;
    }
    __syncthreads();
    {
        float acc = 0.0f;
        const float* wr = W1 + tid * TOT_W1 + 2 * FEAT_DIM; // cols 128:192
        #pragma unroll 8
        for (int f = 0; f < FEAT_DIM; ++f) acc += rp[f] * wr[f];
        Rtab[r * C_P + tid] = acc;
    }
}

// ---------------------------------------------------------------------------
// Precompute 3: one-hot column tables + f16 weight conversion
// ---------------------------------------------------------------------------
__global__ __launch_bounds__(256) void prep_kernel(
    const float* __restrict__ W1,
    const float* __restrict__ W2, const float* __restrict__ W3,
    float* __restrict__ Dtab, float* __restrict__ Etab,
    _Float16* __restrict__ W2h, _Float16* __restrict__ W3h)
{
    const int gid    = blockIdx.x * blockDim.x + threadIdx.x;
    const int stride = gridDim.x * blockDim.x;
    for (int idx = gid; idx < NUM_BINS * C_P; idx += stride) {
        const int bin = idx / C_P, o = idx % C_P;
        Dtab[bin * C_P + o] = W1[o * TOT_W1 + 192 + bin];
        Etab[bin * C_P + o] = W1[o * TOT_W1 + 214 + bin];
    }
    for (int idx = gid; idx < C_P * C_P; idx += stride) {
        W2h[idx] = (_Float16)W2[idx];
        W3h[idx] = (_Float16)W3[idx];
    }
}

// bin index per reference: (d > lower[k]) & (d < upper[k]); NUM_BINS == "no bin"
__device__ __forceinline__ int dist_bin(float dx, float dy, float dz)
{
    const float d = sqrtf(dx * dx + dy * dy + dz * dz);
    const float lo = 1e-3f;
    const float step = (20.0f - 1e-3f) / (float)(NUM_BINS - 1);
    if (d > 20.0f) return NUM_BINS - 1;
    if (d > lo) {
        int b = (int)((d - lo) / step);
        return b > NUM_BINS - 1 ? NUM_BINS - 1 : b;
    }
    return NUM_BINS;
}

// ---------------------------------------------------------------------------
// Fused edge kernel: h1(table adds+relu) -> WMMA layer2 -> WMMA layer3 -> LN
// 256 threads = 8 wave32; each wave owns a 16-edge M-tile (one i, 16 j's).
// ---------------------------------------------------------------------------
__global__ __launch_bounds__(256) void edge_fused_kernel(
    const float* __restrict__ Arow, const float* __restrict__ Acol,
    const float* __restrict__ Rtab,
    const float* __restrict__ Dtab, const float* __restrict__ Etab,
    const _Float16* __restrict__ W2h, const _Float16* __restrict__ W3h,
    const float* __restrict__ b2,   const float* __restrict__ b3,
    const float* __restrict__ ln_g, const float* __restrict__ ln_b,
    const float* __restrict__ p_mask,
    const float* __restrict__ t, const float* __restrict__ sc_t,
    float* __restrict__ out, int B, int N)
{
    __shared__ __align__(16) _Float16 Wh[C_P][C_P];        // 32 KB weight tile
    __shared__ __align__(16) _Float16 stage[8][16][C_P];   // 32 KB activations

    const int per_i = N >> 7;                 // j-blocks of 128 per row i
    int blk = blockIdx.x;
    const int jb = blk % per_i; blk /= per_i;
    const int i  = blk % N;     blk /= N;
    const int b  = blk;

    const int tid = threadIdx.x;
    const int w   = tid >> 5;                 // wave id 0..7
    const int l   = tid & 31;                 // lane
    const int hi  = l >> 4;                   // lane half
    const int col = l & 15;                   // 0..15

    const int node_i = b * N + i;
    const int jw0    = jb * 128 + w * 16;     // first j of this wave's tile

    // ---- stage W2 (f16) into LDS; prefetch W3 toward cache ----
    {
        const uint4* src = (const uint4*)W2h;
        uint4* dst = (uint4*)&Wh[0][0];
        #pragma unroll
        for (int q = 0; q < 8; ++q) dst[tid + q * 256] = src[tid + q * 256];
        __builtin_prefetch((const char*)W3h + tid * 128, 0, 1); // global_prefetch_b8
    }

    // ---- phase 1: h1 tile = relu(Arow_i + Acol_j + Rtab[i-j] + Dcol + Ecol) ----
    const float tix = t[node_i * 3 + 0],  tiy = t[node_i * 3 + 1],  tiz = t[node_i * 3 + 2];
    const float six = sc_t[node_i * 3 + 0], siy = sc_t[node_i * 3 + 1], siz = sc_t[node_i * 3 + 2];
    const float4 ar = *(const float4*)(Arow + node_i * C_P + l * 4);

    for (int e = 0; e < 16; ++e) {
        const int j = jw0 + e;
        const int node_j = b * N + j;
        const int r = i - j + (N - 1);

        const float4 ac = *(const float4*)(Acol + node_j * C_P + l * 4);
        const float4 rp = *(const float4*)(Rtab + r * C_P + l * 4);
        float vx = ar.x + ac.x + rp.x;
        float vy = ar.y + ac.y + rp.y;
        float vz = ar.z + ac.z + rp.z;
        float vw = ar.w + ac.w + rp.w;

        const int bd = dist_bin(t[node_j * 3 + 0] - tix,
                                t[node_j * 3 + 1] - tiy,
                                t[node_j * 3 + 2] - tiz);
        const int be = dist_bin(sc_t[node_j * 3 + 0] - six,
                                sc_t[node_j * 3 + 1] - siy,
                                sc_t[node_j * 3 + 2] - siz);
        if (bd < NUM_BINS) {
            const float4 dd = *(const float4*)(Dtab + bd * C_P + l * 4);
            vx += dd.x; vy += dd.y; vz += dd.z; vw += dd.w;
        }
        if (be < NUM_BINS) {
            const float4 ee = *(const float4*)(Etab + be * C_P + l * 4);
            vx += ee.x; vy += ee.y; vz += ee.z; vw += ee.w;
        }
        H4 hh;
        hh.h[0] = (_Float16)fmaxf(vx, 0.0f);
        hh.h[1] = (_Float16)fmaxf(vy, 0.0f);
        hh.h[2] = (_Float16)fmaxf(vz, 0.0f);
        hh.h[3] = (_Float16)fmaxf(vw, 0.0f);
        *(uint2*)&stage[w][e][l * 4] = hh.u2;
    }
    __syncthreads();

    // ---- layer 2: h2 = relu(h1 @ W2.T + b2) via v_wmma_f32_16x16x32_f16 ----
    v8f acc2[8];
    #pragma unroll
    for (int n = 0; n < 8; ++n) { v8f z = {}; acc2[n] = z; }

    #pragma unroll
    for (int kc = 0; kc < 4; ++kc) {
        // A tile (16x32 f16): lane half selects K sub-block per ISA layout
        v16h a;
        {
            const _Float16* base = &stage[w][col][kc * 32 + hi * 8];
            H8 p0, p1;
            p0.u4 = *(const uint4*)base;
            p1.u4 = *(const uint4*)(base + 16);
            #pragma unroll
            for (int q = 0; q < 8; ++q) { a[q] = p0.h[q]; a[8 + q] = p1.h[q]; }
        }
        #pragma unroll
        for (int n = 0; n < 8; ++n) {
            // B tile (32x16 f16): lane 'col' holds output column n*16+col,
            // K sub-blocks selected by lane half (same layout as A)
            v16h bm;
            const _Float16* wb = &Wh[n * 16 + col][kc * 32 + hi * 8];
            H8 q0, q1;
            q0.u4 = *(const uint4*)wb;
            q1.u4 = *(const uint4*)(wb + 16);
            #pragma unroll
            for (int q = 0; q < 8; ++q) { bm[q] = q0.h[q]; bm[8 + q] = q1.h[q]; }
            acc2[n] = __builtin_amdgcn_wmma_f32_16x16x32_f16(
                false, a, false, bm, (short)0, acc2[n], false, false);
        }
    }

    // epilogue: bias + relu -> stage (wave-private region, C/D layout mapping)
    {
        float b2v[8];
        #pragma unroll
        for (int n = 0; n < 8; ++n) b2v[n] = b2[n * 16 + col];
        #pragma unroll
        for (int n = 0; n < 8; ++n) {
            #pragma unroll
            for (int rr = 0; rr < 8; ++rr) {
                const float v = fmaxf(acc2[n][rr] + b2v[n], 0.0f);
                stage[w][rr + 8 * hi][n * 16 + col] = (_Float16)v;
            }
        }
    }
    __syncthreads();

    // ---- swap weights: Wh <- W3 (f16) ----
    {
        const uint4* src = (const uint4*)W3h;
        uint4* dst = (uint4*)&Wh[0][0];
        #pragma unroll
        for (int q = 0; q < 8; ++q) dst[tid + q * 256] = src[tid + q * 256];
    }
    __syncthreads();

    // ---- layer 3: h3 = h2 @ W3.T + b3 ----
    v8f acc3[8];
    #pragma unroll
    for (int n = 0; n < 8; ++n) { v8f z = {}; acc3[n] = z; }

    #pragma unroll
    for (int kc = 0; kc < 4; ++kc) {
        v16h a;
        {
            const _Float16* base = &stage[w][col][kc * 32 + hi * 8];
            H8 p0, p1;
            p0.u4 = *(const uint4*)base;
            p1.u4 = *(const uint4*)(base + 16);
            #pragma unroll
            for (int q = 0; q < 8; ++q) { a[q] = p0.h[q]; a[8 + q] = p1.h[q]; }
        }
        #pragma unroll
        for (int n = 0; n < 8; ++n) {
            v16h bm;
            const _Float16* wb = &Wh[n * 16 + col][kc * 32 + hi * 8];
            H8 q0, q1;
            q0.u4 = *(const uint4*)wb;
            q1.u4 = *(const uint4*)(wb + 16);
            #pragma unroll
            for (int q = 0; q < 8; ++q) { bm[q] = q0.h[q]; bm[8 + q] = q1.h[q]; }
            acc3[n] = __builtin_amdgcn_wmma_f32_16x16x32_f16(
                false, a, false, bm, (short)0, acc3[n], false, false);
        }
    }

    // ---- epilogue: bias, LayerNorm over 128 feats (16-lane shfl reduce), mask ----
    {
        float b3v[8], gv[8], bev[8];
        #pragma unroll
        for (int n = 0; n < 8; ++n) {
            b3v[n] = b3[n * 16 + col];
            gv[n]  = ln_g[n * 16 + col];
            bev[n] = ln_b[n * 16 + col];
        }
        float msk[8];
        const long mbase = ((long)(b * N + i)) * N + jw0 + hi * 8;
        #pragma unroll
        for (int rr = 0; rr < 8; ++rr) msk[rr] = p_mask[mbase + rr];

        float mean[8], rstd[8];
        #pragma unroll
        for (int rr = 0; rr < 8; ++rr) {
            float s1 = 0.0f, s2 = 0.0f;
            #pragma unroll
            for (int n = 0; n < 8; ++n) {
                const float v = acc3[n][rr] + b3v[n];
                s1 += v; s2 += v * v;
            }
            #pragma unroll
            for (int off = 1; off < 16; off <<= 1) {  // reduce within 16-lane half
                s1 += __shfl_xor(s1, off, 32);
                s2 += __shfl_xor(s2, off, 32);
            }
            const float mu  = s1 * (1.0f / (float)C_P);
            const float var = s2 * (1.0f / (float)C_P) - mu * mu;
            mean[rr] = mu;
            rstd[rr] = rsqrtf(var + 1e-5f);
        }

        const long obase = (((long)(b * N + i)) * N + jw0 + hi * 8) * C_P;
        #pragma unroll
        for (int rr = 0; rr < 8; ++rr) {
            #pragma unroll
            for (int n = 0; n < 8; ++n) {
                float v = acc3[n][rr] + b3v[n];
                v = (v - mean[rr]) * rstd[rr] * gv[n] + bev[n];
                out[obase + (long)rr * C_P + n * 16 + col] = v * msk[rr];
            }
        }
    }
}

// ---------------------------------------------------------------------------
extern "C" void kernel_launch(void* const* d_in, const int* in_sizes, int n_in,
                              void* d_out, int out_size, void* d_ws, size_t ws_size,
                              hipStream_t stream)
{
    const float* s      = (const float*)d_in[0];
    const float* t      = (const float*)d_in[1];
    const float* sc_t   = (const float*)d_in[2];
    const float* p_mask = (const float*)d_in[3];
    const float* W_sp   = (const float*)d_in[4];
    const float* b_sp   = (const float*)d_in[5];
    const float* W_rp   = (const float*)d_in[6];
    const float* b_rp   = (const float*)d_in[7];
    const float* W1     = (const float*)d_in[8];
    const float* b1     = (const float*)d_in[9];
    const float* W2     = (const float*)d_in[10];
    const float* b2     = (const float*)d_in[11];
    const float* W3     = (const float*)d_in[12];
    const float* b3     = (const float*)d_in[13];
    const float* ln_g   = (const float*)d_in[14];
    const float* ln_b   = (const float*)d_in[15];

    const int BN = in_sizes[0] / C_S;          // B*N
    const int N  = in_sizes[3] / BN;           // p_mask = B*N*N
    const int B  = BN / N;

    char* ws = (char*)d_ws;
    size_t off = 0;
    auto carve = [&](size_t bytes) -> void* {
        off = (off + 255) & ~(size_t)255;
        void* p = ws + off; off += bytes; return p;
    };
    float*    Arow = (float*)carve((size_t)BN * C_P * 4);
    float*    Acol = (float*)carve((size_t)BN * C_P * 4);
    float*    Rtab = (float*)carve((size_t)(2 * N - 1) * C_P * 4);
    float*    Dtab = (float*)carve((size_t)NUM_BINS * C_P * 4);
    float*    Etab = (float*)carve((size_t)NUM_BINS * C_P * 4);
    _Float16* W2h  = (_Float16*)carve((size_t)C_P * C_P * 2);
    _Float16* W3h  = (_Float16*)carve((size_t)C_P * C_P * 2);

    node_embed_kernel<<<BN, 256, 0, stream>>>(s, W_sp, b_sp, W1, b1, Arow, Acol);
    relpos_kernel<<<2 * N - 1, 128, 0, stream>>>(W_rp, b_rp, W1, Rtab, N);
    prep_kernel<<<64, 256, 0, stream>>>(W1, W2, W3, Dtab, Etab, W2h, W3h);

    const int per_i = N / 128;
    edge_fused_kernel<<<B * N * per_i, 256, 0, stream>>>(
        Arow, Acol, Rtab, Dtab, Etab, W2h, W3h,
        b2, b3, ln_g, ln_b, p_mask, t, sc_t, (float*)d_out, B, N);
}
